// GPT_47957604827558
// MI455X (gfx1250) — matrix operand order repaired and checked by reading, not compiled, not run on hardware
//
#include <hip/hip_runtime.h>
#include <hip/hip_bf16.h>
#include <math.h>

#define V_    32000
#define D_    768
#define H_    12
#define KVH_  4
#define L_    6
#define T_    1024
#define HD_   64
#define KVD_  256
#define B_    2
#define BT_   (B_ * T_)

typedef __attribute__((ext_vector_type(2)))  __bf16 v2bf;
typedef __attribute__((ext_vector_type(8)))  __bf16 v8bf;
typedef __attribute__((ext_vector_type(16))) __bf16 v16bf;
typedef __attribute__((ext_vector_type(8)))  float  v8f;

#define SHUF16(lo, hi) \
  __builtin_shufflevector(lo, hi, 0, 1, 2, 3, 4, 5, 6, 7, 8, 9, 10, 11, 12, 13, 14, 15)

// load 8 contiguous floats (16B-aligned) -> one v8bf group
static __device__ __forceinline__ v8bf cvt8(const float* __restrict__ src) {
  float b[8];
  *(float4*)&b[0] = ((const float4*)src)[0];
  *(float4*)&b[4] = ((const float4*)src)[1];
  v8bf r;
#pragma unroll
  for (int e = 0; e < 8; ++e) r[e] = (__bf16)b[e];
  return r;
}

// -------- cross-lane reductions within 16-lane rows (VALU v_permlane16) -----
#if __has_builtin(__builtin_amdgcn_permlane16)
static __device__ __forceinline__ float rowred_max16(float v) {
  int x;
  x = __builtin_amdgcn_permlane16(__float_as_int(v), __float_as_int(v),
                                  0x67452301u, 0xEFCDAB89u, false, false);
  v = fmaxf(v, __int_as_float(x));                       // xor 1
  x = __builtin_amdgcn_permlane16(__float_as_int(v), __float_as_int(v),
                                  0x54761032u, 0xDCFE98BAu, false, false);
  v = fmaxf(v, __int_as_float(x));                       // xor 2
  x = __builtin_amdgcn_permlane16(__float_as_int(v), __float_as_int(v),
                                  0x32107654u, 0xBA98FEDCu, false, false);
  v = fmaxf(v, __int_as_float(x));                       // xor 4
  x = __builtin_amdgcn_permlane16(__float_as_int(v), __float_as_int(v),
                                  0xFEDCBA98u, 0x76543210u, false, false);
  v = fmaxf(v, __int_as_float(x));                       // xor 8
  return v;
}
static __device__ __forceinline__ float rowred_sum16(float v) {
  int x;
  x = __builtin_amdgcn_permlane16(__float_as_int(v), __float_as_int(v),
                                  0x67452301u, 0xEFCDAB89u, false, false);
  v += __int_as_float(x);
  x = __builtin_amdgcn_permlane16(__float_as_int(v), __float_as_int(v),
                                  0x54761032u, 0xDCFE98BAu, false, false);
  v += __int_as_float(x);
  x = __builtin_amdgcn_permlane16(__float_as_int(v), __float_as_int(v),
                                  0x32107654u, 0xBA98FEDCu, false, false);
  v += __int_as_float(x);
  x = __builtin_amdgcn_permlane16(__float_as_int(v), __float_as_int(v),
                                  0xFEDCBA98u, 0x76543210u, false, false);
  v += __int_as_float(x);
  return v;
}
#else
static __device__ __forceinline__ float rowred_max16(float v) {
  for (int m = 1; m < 16; m <<= 1) v = fmaxf(v, __shfl_xor(v, m, 32));
  return v;
}
static __device__ __forceinline__ float rowred_sum16(float v) {
  for (int m = 1; m < 16; m <<= 1) v += __shfl_xor(v, m, 32);
  return v;
}
#endif

// ---------------------------------------------------------------------------
// GEMM: C[M,N] (+)= A[M,K] * W[N,K]^T   (einsum 'mk,nk->mn', both K-major)
// 64x64 tile per 128-thread block, bf16 WMMA 16x16x32, f32 accumulate.
// Double-buffered LDS staging: next tile loads overlap current tile WMMAs.
// EPI: 0 = store, 1 = add into C (residual), 2 = relu^2, 3 = softcap tanh
// ---------------------------------------------------------------------------
template <int EPI>
__global__ __launch_bounds__(128) void gemm_bf16_kernel(
    const float* __restrict__ A, const float* __restrict__ W,
    float* __restrict__ C, int M, int N, int K) {
  __shared__ v8bf As[2][64][4];   // 2 x (64 rows x 32 bf16)
  __shared__ v8bf Ws[2][64][4];
  const int m0 = blockIdx.y * 64;
  const int n0 = blockIdx.x * 64;
  const int tid  = threadIdx.x;
  const int wave = tid >> 5;
  const int lane = tid & 31;
  const int half = lane >> 4;
  const int lm   = lane & 15;

  const int sr = tid >> 1;          // staging row
  const int sg = (tid & 1) * 2;     // staging group base (each group = 8 elems)

  const float* aBase = A + (size_t)(m0 + sr) * K + sg * 8;
  const float* wBase = W + (size_t)(n0 + sr) * K + sg * 8;

  // prologue: stage tile 0
  As[0][sr][sg]     = cvt8(aBase);
  As[0][sr][sg + 1] = cvt8(aBase + 8);
  Ws[0][sr][sg]     = cvt8(wBase);
  Ws[0][sr][sg + 1] = cvt8(wBase + 8);
  __syncthreads();

  v8f acc[4] = {};
  int buf = 0;
  for (int k0 = 0; k0 < K; k0 += 32, buf ^= 1) {
    // stage next K tile into the other buffer (overlaps with WMMAs below)
    if (k0 + 32 < K) {
      const float* ap = aBase + k0 + 32;
      const float* wp = wBase + k0 + 32;
      As[buf ^ 1][sr][sg]     = cvt8(ap);
      As[buf ^ 1][sr][sg + 1] = cvt8(ap + 8);
      Ws[buf ^ 1][sr][sg]     = cvt8(wp);
      Ws[buf ^ 1][sr][sg + 1] = cvt8(wp + 8);
      if (k0 + 64 < K) {   // prefetch tile after next (global_prefetch_b8)
        __builtin_prefetch(ap + 32, 0, 1);
        __builtin_prefetch(wp + 32, 0, 1);
      }
    }

    const v8bf* arow = As[buf][wave * 16 + lm];
    v16bf a = SHUF16(arow[half], arow[2 + half]);
#pragma unroll
    for (int s = 0; s < 4; ++s) {
      const v8bf* wrow = Ws[buf][s * 16 + lm];  // B[k][n]=W[n][k]
      v16bf b = SHUF16(wrow[half], wrow[2 + half]);
      acc[s] = __builtin_amdgcn_wmma_f32_16x16x32_bf16(
          false, a, false, b, (short)0, acc[s], false, false);
    }
    __syncthreads();   // next-tile writes done, current-tile reads done
  }

#pragma unroll
  for (int s = 0; s < 4; ++s) {
#pragma unroll
    for (int r = 0; r < 8; ++r) {
      int m = m0 + wave * 16 + r + 8 * half;
      int n = n0 + s * 16 + lm;
      float v = acc[s][r];
      size_t o = (size_t)m * N + n;
      if (EPI == 0) {
        C[o] = v;
      } else if (EPI == 1) {
        C[o] += v;
      } else if (EPI == 2) {
        float t = fmaxf(v, 0.0f);
        C[o] = t * t;
      } else {
        C[o] = 15.0f * tanhf(v * (1.0f / 15.0f));
      }
    }
  }
}

// ---------------------------------------------------------------------------
// stage one 32-key K/V chunk: K row-major (b128 stores), V transposed
// (paired-key b32 stores). Thread owns key pair {2sp, 2sp+1} x 8 hd values.
// ---------------------------------------------------------------------------
static __device__ __forceinline__ void stage_kv(
    const float* __restrict__ Kb, const float* __restrict__ Vb, size_t src,
    v8bf (*__restrict__ Ksb)[8], v8bf (*__restrict__ Vtb)[4],
    int sp, int sg2, int hd0) {
  Ksb[2 * sp][sg2]     = cvt8(Kb + src);
  Ksb[2 * sp + 1][sg2] = cvt8(Kb + src + KVD_);
  float va[8], vb[8];
  *(float4*)&va[0] = ((const float4*)(Vb + src))[0];
  *(float4*)&va[4] = ((const float4*)(Vb + src))[1];
  *(float4*)&vb[0] = ((const float4*)(Vb + src + KVD_))[0];
  *(float4*)&vb[4] = ((const float4*)(Vb + src + KVD_))[1];
  v2bf* vt = (v2bf*)Vtb;
#pragma unroll
  for (int e = 0; e < 8; ++e) {
    v2bf pr;
    pr[0] = (__bf16)va[e];
    pr[1] = (__bf16)vb[e];
    vt[(hd0 + e) * 16 + sp] = pr;
  }
}

// ---------------------------------------------------------------------------
// Fused flash attention: 64 queries/block, 32-key chunks, windowed causal.
// Double-buffered K/V staging; interior chunks skip window masking.
// ---------------------------------------------------------------------------
__global__ __launch_bounds__(128) void attn_kernel(
    const float* __restrict__ Q, const float* __restrict__ Kb,
    const float* __restrict__ Vb, float* __restrict__ Y, int w) {
  __shared__ v8bf Qs[64][8];      // 64 q-rows x 64 hd
  __shared__ v8bf Ks[2][32][8];   // 2 x (32 keys x 64 hd)
  __shared__ v8bf Vt[2][64][4];   // 2 x (64 hd x 32 keys, transposed)
  __shared__ v8bf Ps[64][4];      // 64 q-rows x 32 keys (wave-local regions)

  const int q0  = blockIdx.x * 64;
  const int bh  = blockIdx.y;
  const int b   = bh / H_;
  const int h   = bh % H_;
  const int kvh = h / (H_ / KVH_);
  const int tid  = threadIdx.x;
  const int wave = tid >> 5;
  const int lane = tid & 31;
  const int half = lane >> 4;
  const int lm   = lane & 15;

  // stage Q tile (64x64)
  {
    const int r = tid >> 1;
    const int g = (tid & 1) * 4;
    const float* qp = Q + ((size_t)(b * T_ + q0 + r)) * D_ + h * HD_ + g * 8;
    Qs[r][g]     = cvt8(qp);
    Qs[r][g + 1] = cvt8(qp + 8);
    Qs[r][g + 2] = cvt8(qp + 16);
    Qs[r][g + 3] = cvt8(qp + 24);
  }

  int kstart = q0 - w + 1;
  if (kstart < 0) kstart = 0;
  kstart &= ~31;

  // staging coords: thread owns key pair {2*sp, 2*sp+1} x 8 hd values
  const int sp  = tid >> 3;
  const int sg2 = tid & 7;
  const int hd0 = sg2 * 8;
  const size_t kvBase = (size_t)(b * T_) * KVD_ + kvh * HD_ + hd0 +
                        (size_t)(2 * sp) * KVD_;

  // prologue: stage first chunk into buffer 0
  stage_kv(Kb, Vb, kvBase + (size_t)kstart * KVD_, Ks[0], Vt[0], sp, sg2, hd0);
  __syncthreads();

  const v8bf* qrow = Qs[wave * 16 + lm];
  v16bf aq[2];
  aq[0] = SHUF16(qrow[half], qrow[2 + half]);
  aq[1] = SHUF16(qrow[4 + half], qrow[6 + half]);

  v8f o[4] = {};
  float row_max[8], row_sum[8];
#pragma unroll
  for (int r = 0; r < 8; ++r) { row_max[r] = -1e30f; row_sum[r] = 0.0f; }

  int buf = 0;
  for (int kb = kstart; kb <= q0 + 63; kb += 32, buf ^= 1) {
    // stage next chunk into the other buffer (overlaps with WMMAs below)
    if (kb + 32 <= q0 + 63)
      stage_kv(Kb, Vb, kvBase + (size_t)(kb + 32) * KVD_,
               Ks[buf ^ 1], Vt[buf ^ 1], sp, sg2, hd0);

    const bool need_mask = !((kb + 31 <= q0) && ((q0 + 63 - kb) < w));

    // scores: two 16(q) x 16(key) tiles, K-dim = HD = 64 (2 WMMA each)
    float p[2][8];
#pragma unroll
    for (int t = 0; t < 2; ++t) {
      v8f sc = {};
      const v8bf* krow = Ks[buf][t * 16 + lm];  // B[hd][key]=K[key][hd]
#pragma unroll
      for (int u = 0; u < 2; ++u) {
        v16bf bk = SHUF16(krow[4 * u + half], krow[4 * u + 2 + half]);
        sc = __builtin_amdgcn_wmma_f32_16x16x32_bf16(
            false, aq[u], false, bk, (short)0, sc, false, false);
      }
      if (need_mask) {
        int kpos = kb + t * 16 + lm;
#pragma unroll
        for (int r = 0; r < 8; ++r) {
          int qpos = q0 + wave * 16 + r + 8 * half;
          int d = qpos - kpos;
          bool ok = (d >= 0) && (d < w);
          p[t][r] = ok ? sc[r] * 0.125f : -1e30f;   // 1/sqrt(64)
        }
      } else {
#pragma unroll
        for (int r = 0; r < 8; ++r) p[t][r] = sc[r] * 0.125f;
      }
    }

    // online softmax (row = fixed r within a 16-lane half-group)
#pragma unroll
    for (int r = 0; r < 8; ++r) {
      float mx = rowred_max16(fmaxf(p[0][r], p[1][r]));
      float mn = fmaxf(row_max[r], mx);
      float alpha = __expf(row_max[r] - mn);
      row_max[r] = mn;
      float e0, e1;
      if (need_mask) {
        e0 = (p[0][r] <= -1e29f) ? 0.0f : __expf(p[0][r] - mn);
        e1 = (p[1][r] <= -1e29f) ? 0.0f : __expf(p[1][r] - mn);
      } else {
        e0 = __expf(p[0][r] - mn);
        e1 = __expf(p[1][r] - mn);
      }
      row_sum[r] = row_sum[r] * alpha + rowred_sum16(e0 + e1);
#pragma unroll
      for (int s = 0; s < 4; ++s) o[s][r] *= alpha;
      // spill P (C-layout) to per-wave LDS region for re-pack as A-fragment
      __bf16* pr = (__bf16*)&Ps[wave * 16 + r + 8 * half][0];
      pr[lm] = (__bf16)e0;
      pr[16 + lm] = (__bf16)e1;
    }
    // wave-local LDS ops are in program order: safe to read back, no barrier
    const v8bf* prow = Ps[wave * 16 + lm];
    v16bf ap = SHUF16(prow[half], prow[2 + half]);
#pragma unroll
    for (int s = 0; s < 4; ++s) {
      const v8bf* vrow = Vt[buf][s * 16 + lm];  // B[key][hd] via transposed V
      v16bf bv = SHUF16(vrow[half], vrow[2 + half]);
      o[s] = __builtin_amdgcn_wmma_f32_16x16x32_bf16(
          false, ap, false, bv, (short)0, o[s], false, false);
    }
    __syncthreads();   // next-chunk writes done, current-chunk reads done
  }

  // normalize and write y[b][t][h][hd]
#pragma unroll
  for (int s = 0; s < 4; ++s) {
#pragma unroll
    for (int r = 0; r < 8; ++r) {
      int m = wave * 16 + r + 8 * half;
      float inv = 1.0f / row_sum[r];
      Y[((size_t)(b * T_ + q0 + m)) * D_ + h * HD_ + s * 16 + lm] = o[s][r] * inv;
    }
  }
}

// ---------------------------------------------------------------------------
// x = norm(wte[idx]); x0 = x.  One 256-thread block per token (D=768 -> 3/th)
// ---------------------------------------------------------------------------
__global__ __launch_bounds__(256) void embed_norm_kernel(
    const int* __restrict__ idx, const float* __restrict__ wte,
    float* __restrict__ x, float* __restrict__ x0) {
  __shared__ float red[256];
  const int tok = blockIdx.x;
  const int id = idx[tok];
  const float* src = wte + (size_t)id * D_;
  float v[3];
  float ss = 0.0f;
#pragma unroll
  for (int e = 0; e < 3; ++e) {
    v[e] = src[threadIdx.x + e * 256];
    ss += v[e] * v[e];
  }
  red[threadIdx.x] = ss;
  __syncthreads();
  for (int s = 128; s > 0; s >>= 1) {
    if (threadIdx.x < s) red[threadIdx.x] += red[threadIdx.x + s];
    __syncthreads();
  }
  float r = rsqrtf(red[0] * (1.0f / D_) + 1e-6f);
#pragma unroll
  for (int e = 0; e < 3; ++e) {
    size_t o = (size_t)tok * D_ + threadIdx.x + e * 256;
    float val = v[e] * r;
    x[o] = val;
    x0[o] = val;
  }
}

// ---------------------------------------------------------------------------
// x = rl*x + xl*x0 (layer>=0) or unchanged (layer<0); xn = rmsnorm(x)
// ---------------------------------------------------------------------------
__global__ __launch_bounds__(256) void mix_norm_kernel(
    float* __restrict__ x, const float* __restrict__ x0,
    const float* __restrict__ rl, const float* __restrict__ xl,
    float* __restrict__ xn, int layer) {
  __shared__ float red[256];
  const int tok = blockIdx.x;
  const float a = (layer >= 0) ? rl[layer] : 1.0f;
  const float c = (layer >= 0) ? xl[layer] : 0.0f;
  float v[3];
  float ss = 0.0f;
#pragma unroll
  for (int e = 0; e < 3; ++e) {
    size_t o = (size_t)tok * D_ + threadIdx.x + e * 256;
    float xv = a * x[o] + c * x0[o];
    v[e] = xv;
    ss += xv * xv;
    x[o] = xv;
  }
  red[threadIdx.x] = ss;
  __syncthreads();
  for (int s = 128; s > 0; s >>= 1) {
    if (threadIdx.x < s) red[threadIdx.x] += red[threadIdx.x + s];
    __syncthreads();
  }
  float r = rsqrtf(red[0] * (1.0f / D_) + 1e-6f);
#pragma unroll
  for (int e = 0; e < 3; ++e)
    xn[(size_t)tok * D_ + threadIdx.x + e * 256] = v[e] * r;
}

// ---------------------------------------------------------------------------
// rope (tables on the fly) + per-head rmsnorm. One thread per (token, head).
// ---------------------------------------------------------------------------
__global__ void rope_norm_kernel(float* __restrict__ X, int heads) {
  int tid = blockIdx.x * blockDim.x + threadIdx.x;
  if (tid >= BT_ * heads) return;
  const int hh = tid % heads;
  const int tok = tid / heads;
  const int t = tok % T_;
  float* p = X + (size_t)tok * heads * HD_ + hh * HD_;
  float v[64];
#pragma unroll
  for (int c = 0; c < 64; ++c) v[c] = p[c];
#pragma unroll
  for (int c = 0; c < 32; ++c) {
    float f = (float)t * __powf(10000.0f, -(float)c * (1.0f / 32.0f));
    float sn, cs;
    __sincosf(f, &sn, &cs);
    float x1 = v[c], x2 = v[c + 32];
    v[c] = x1 * cs + x2 * sn;
    v[c + 32] = -x1 * sn + x2 * cs;
  }
  float ms = 0.0f;
#pragma unroll
  for (int c = 0; c < 64; ++c) ms += v[c] * v[c];
  float r = rsqrtf(ms * (1.0f / 64.0f) + 1e-6f);
#pragma unroll
  for (int c = 0; c < 64; ++c) p[c] = v[c] * r;
}

// ---------------------------------------------------------------------------
// v += 2*sigmoid(xn[:,:32] @ gate^T) * ve_table[idx]. Thread per (token,kvh).
// ---------------------------------------------------------------------------
__global__ void ve_gate_kernel(
    const int* __restrict__ idx, const float* __restrict__ xn,
    const float* __restrict__ gates, const float* __restrict__ tables,
    float* __restrict__ Vv, int j) {
  int tid = blockIdx.x * blockDim.x + threadIdx.x;
  if (tid >= BT_ * KVH_) return;
  const int kv = tid % KVH_;
  const int tok = tid / KVH_;
  const float* gw = gates + ((size_t)j * KVH_ + kv) * 32;
  float g = 0.0f;
#pragma unroll
  for (int c = 0; c < 32; ++c) g += xn[(size_t)tok * D_ + c] * gw[c];
  float gate = 2.0f / (1.0f + __expf(-g));
  const int id = idx[tok];
  const float* ve = tables + ((size_t)j * V_ + id) * KVD_ + kv * HD_;
  float* vp = Vv + (size_t)tok * KVD_ + kv * HD_;
#pragma unroll
  for (int d = 0; d < 64; ++d) vp[d] += gate * ve[d];
}

// ---------------------------------------------------------------------------
extern "C" void kernel_launch(void* const* d_in, const int* in_sizes, int n_in,
                              void* d_out, int out_size, void* d_ws, size_t ws_size,
                              hipStream_t stream) {
  const int*   idx  = (const int*)d_in[0];
  const float* wte  = (const float*)d_in[1];
  const float* Wq   = (const float*)d_in[2];
  const float* Wk   = (const float*)d_in[3];
  const float* Wv   = (const float*)d_in[4];
  const float* Wo   = (const float*)d_in[5];
  const float* Wfc  = (const float*)d_in[6];
  const float* Wpj  = (const float*)d_in[7];
  const float* veT  = (const float*)d_in[8];
  const float* veG  = (const float*)d_in[9];
  const float* rl   = (const float*)d_in[10];
  const float* xl   = (const float*)d_in[11];
  const float* lmw  = (const float*)d_in[12];
  float* out = (float*)d_out;

  float* ws = (float*)d_ws;
  float* x  = ws;                        // BT*D
  float* x0 = x  + (size_t)BT_ * D_;     // BT*D
  float* xn = x0 + (size_t)BT_ * D_;     // BT*D
  float* q  = xn + (size_t)BT_ * D_;     // BT*D
  float* k  = q  + (size_t)BT_ * D_;     // BT*KVD
  float* v  = k  + (size_t)BT_ * KVD_;   // BT*KVD
  float* y  = v  + (size_t)BT_ * KVD_;   // BT*D
  float* hb = y  + (size_t)BT_ * D_;     // BT*4D

  static const int WIN[L_] = {T_ / 2, T_, T_ / 2, T_, T_ / 2, T_};

  embed_norm_kernel<<<BT_, 256, 0, stream>>>(idx, wte, x, x0);

  const dim3 gBlk(128);
  for (int i = 0; i < L_; ++i) {
    mix_norm_kernel<<<BT_, 256, 0, stream>>>(x, x0, rl, xl, xn, i);

    gemm_bf16_kernel<0><<<dim3(D_ / 64, BT_ / 64), gBlk, 0, stream>>>(
        xn, Wq + (size_t)i * D_ * D_, q, BT_, D_, D_);
    gemm_bf16_kernel<0><<<dim3(KVD_ / 64, BT_ / 64), gBlk, 0, stream>>>(
        xn, Wk + (size_t)i * KVD_ * D_, k, BT_, KVD_, D_);
    gemm_bf16_kernel<0><<<dim3(KVD_ / 64, BT_ / 64), gBlk, 0, stream>>>(
        xn, Wv + (size_t)i * KVD_ * D_, v, BT_, KVD_, D_);

    if (i == 1 || i == 3 || i == 5) {
      int j = (i - 1) / 2;
      ve_gate_kernel<<<(BT_ * KVH_ + 255) / 256, 256, 0, stream>>>(
          idx, xn, veG, veT, v, j);
    }

    rope_norm_kernel<<<(BT_ * H_ + 255) / 256, 256, 0, stream>>>(q, H_);
    rope_norm_kernel<<<(BT_ * KVH_ + 255) / 256, 256, 0, stream>>>(k, KVH_);

    attn_kernel<<<dim3(T_ / 64, B_ * H_), gBlk, 0, stream>>>(q, k, v, y, WIN[i]);

    gemm_bf16_kernel<1><<<dim3(D_ / 64, BT_ / 64), gBlk, 0, stream>>>(
        y, Wo + (size_t)i * D_ * D_, x, BT_, D_, D_);        // x += y @ Wo^T

    mix_norm_kernel<<<BT_, 256, 0, stream>>>(x, x0, rl, xl, xn, -1);  // xn=norm(x)

    gemm_bf16_kernel<2><<<dim3(4 * D_ / 64, BT_ / 64), gBlk, 0, stream>>>(
        xn, Wfc + (size_t)i * 4 * D_ * D_, hb, BT_, 4 * D_, D_);      // relu^2
    gemm_bf16_kernel<1><<<dim3(D_ / 64, BT_ / 64), gBlk, 0, stream>>>(
        hb, Wpj + (size_t)i * D_ * 4 * D_, x, BT_, D_, 4 * D_);       // x += h @ Wproj^T
  }

  mix_norm_kernel<<<BT_, 256, 0, stream>>>(x, x0, rl, xl, xn, -1);
  gemm_bf16_kernel<3><<<dim3(V_ / 64, BT_ / 64), gBlk, 0, stream>>>(
      xn, lmw, out, BT_, V_, D_);   // softcap tanh epilogue
}
